// LSTM_23845658428366
// MI455X (gfx1250) — matrix operand order repaired and checked by reading, not compiled, not run on hardware
//
#include <hip/hip_runtime.h>
#include <hip/hip_bf16.h>

// ---------------------------------------------------------------------------
// Persistent-kernel LSTM for MI455X (gfx1250, wave32, WMMA).
//   B=32, T=2048, D=256, H=512, O=256.
// 32 workgroups; WG g owns h-columns [16g,16g+16) and the matching 4 gate
// column groups. Weight slices live in LDS as bf16 WMMA B-fragments for the
// whole scan. One device-wide barrier per timestep; double-buffered h.
//
// Fast path (needs ~32MB ws): X is pre-converted to bf16 in WMMA A-fragment
// layout once; h lives in a fragment-layout double buffer. Per step, waves
// load A fragments straight from global (b128/lane) -> WMMA, no LDS staging.
// Fallback path (128KB ws): stage [x_t | h] through LDS each step.
// ---------------------------------------------------------------------------

typedef __attribute__((ext_vector_type(16))) __bf16 v16bf_t;
typedef __attribute__((ext_vector_type(8)))  float  v8f_t;

namespace {
constexpr int kB = 32, kT = 2048, kD = 256, kH = 512, kO = 256;
constexpr int kFourH = 4 * kH;           // 2048
constexpr int kKtot  = kD + kH;          // 768 fused K ([x | h] @ [Wx; Wh])
constexpr int kKC    = kKtot / 32;       // 24 WMMA K-chunks per step
constexpr int kKCX   = kD / 32;          // 8 x-chunks
constexpr int kKCH   = kH / 32;          // 16 h-chunks
constexpr int kNWG   = kH / 16;          // 32 persistent workgroups
constexpr int kNT    = 256;              // 8 waves per WG

// ---- fragment-path workspace layout (global) ----
// One fragment group = [32 lanes][16 bf16] = 1KB.
constexpr size_t kFragElems   = 32 * 16;                        // 512 bf16
constexpr size_t kXFPerT      = (size_t)2 * kKCX * kFragElems;  // 8192 bf16 / t
constexpr size_t kXFBytes     = (size_t)kT * kXFPerT * 2;       // 32 MB
constexpr size_t kHFPerBuf    = (size_t)2 * kKCH * kFragElems;  // 16384 bf16
constexpr size_t kHFBytes     = 2 * kHFPerBuf * 2;              // 64 KB
constexpr size_t kOffHF       = 256;
constexpr size_t kOffXF       = kOffHF + kHFBytes;              // 65792 (256-aligned)
constexpr size_t kNeedFrag    = kOffXF + kXFBytes;

// ---- fragment-path LDS ----
constexpr size_t kF_OffBW = 0;
constexpr size_t kF_SzBW  = (size_t)4 * kKC * 32 * 16 * sizeof(__bf16);   // 98304
constexpr size_t kF_OffZB = kF_OffBW + kF_SzBW;
constexpr size_t kF_SzZB  = (size_t)8 * 32 * 8 * sizeof(float);           // 8192
constexpr size_t kF_OffCB = kF_OffZB + kF_SzZB;
constexpr size_t kF_SzCB  = (size_t)2 * 32 * 8 * sizeof(float);           // 2048
constexpr size_t kF_Smem  = kF_OffCB + kF_SzCB;                           // ~106 KB

// ---- staged-path (fallback) LDS ----
constexpr int    kKPad  = kKtot + 8;
constexpr size_t kS_OffBW = 0;
constexpr size_t kS_SzBW  = kF_SzBW;
constexpr size_t kS_OffAS = kS_OffBW + kS_SzBW;
constexpr size_t kS_SzAS  = (size_t)kB * kKPad * sizeof(__bf16);
constexpr size_t kS_OffZB = (kS_OffAS + kS_SzAS + 255) & ~(size_t)255;
constexpr size_t kS_OffCB = kS_OffZB + kF_SzZB;
constexpr size_t kS_Smem  = kS_OffCB + kF_SzCB;                           // ~154.5 KB
constexpr size_t kNeedStaged = 256 + 2 * (size_t)kB * kH * sizeof(float);
} // namespace

__device__ __forceinline__ float sigf(float x)      { return 1.0f / (1.0f + __expf(-x)); }
__device__ __forceinline__ float tanh_fast(float x) { return 2.0f * sigf(2.0f * x) - 1.0f; }

// Release/acquire grid barrier over kNWG blocks; `target` = #arrivals to wait.
__device__ __forceinline__ void grid_barrier(unsigned* bar, unsigned target, int tid) {
  __threadfence();
  __syncthreads();
  if (tid == 0) {
    __hip_atomic_fetch_add(bar, 1u, __ATOMIC_RELEASE, __HIP_MEMORY_SCOPE_AGENT);
    while (__hip_atomic_load(bar, __ATOMIC_ACQUIRE, __HIP_MEMORY_SCOPE_AGENT) < target)
      __builtin_amdgcn_s_sleep(2);
    __threadfence();
  }
  __syncthreads();
}

// Pack [Wx;Wh] gate slices for WG `wg` into LDS bf16 B-fragments.
// B layout (16x16x32): lane n<16 -> col n, elems = chunk K 0..7,16..23;
//                      lane n+16 -> col n, elems = chunk K 8..15,24..31.
__device__ __forceinline__ void pack_weights(
    __bf16 (*bW)[kKC][32][16], const float* __restrict__ Wx,
    const float* __restrict__ Wh, int wg, int tid) {
  const int j0 = wg * 16;
  for (int idx = tid; idx < 4 * kKC * 32; idx += kNT) {
    const int q   = idx / (kKC * 32);
    const int kk  = (idx / 32) % kKC;
    const int ln  = idx & 31;
    const int col = q * kH + j0 + (ln & 15);
    const int kb  = kk * 32 + ((ln & 16) ? 8 : 0);
    #pragma unroll
    for (int e = 0; e < 16; ++e) {
      const int k = kb + e + (e & 8);
      const float w = (k < kD) ? Wx[(size_t)k * kFourH + col]
                               : Wh[(size_t)(k - kD) * kFourH + col];
      bW[q][kk][ln][e] = (__bf16)w;
    }
  }
}

// ===========================================================================
// Fast path: fragment-direct persistent kernel.
// ===========================================================================
__global__ __launch_bounds__(kNT, 1)
void lstm_frag(const float* __restrict__ X,    // [B,T,D]
               const float* __restrict__ Wx,   // [D,4H]
               const float* __restrict__ Wh,   // [H,4H]
               const float* __restrict__ bias, // [4H]
               const float* __restrict__ Wd,   // [H,O]
               const float* __restrict__ bd,   // [O]
               float*  __restrict__ out,       // [B,1,O]
               __bf16* __restrict__ xf,        // [T][2][8][32][16] bf16 A-frags
               __bf16* __restrict__ hf,        // [2][2][16][32][16] bf16 (zeroed)
               unsigned* __restrict__ bar)     // grid barrier counter (zeroed)
{
  extern __shared__ char smem[];
  __bf16 (*bW)[kKC][32][16] = reinterpret_cast<__bf16 (*)[kKC][32][16]>(smem + kF_OffBW);
  float  (*zb)[32][8]       = reinterpret_cast<float (*)[32][8]>(smem + kF_OffZB);
  float  (*cb)[32][8]       = reinterpret_cast<float (*)[32][8]>(smem + kF_OffCB);

  const int wg    = blockIdx.x;       // 0..31 : owns h-cols [16*wg, 16*wg+16)
  const int tid   = threadIdx.x;
  const int lane  = tid & 31;
  const int wave  = tid >> 5;         // 0..7
  const int mt    = wave >> 2;        // M-tile (batch rows 16*mt..16*mt+15)
  const int gq    = wave & 3;         // gate: 0=i 1=f 2=g 3=o
  const int j0    = wg * 16;
  const int nloc  = lane & 15;
  const int hi16  = (lane & 16) ? 1 : 0;

  // ---- Phase 0a: pre-fragment all of X to bf16 (whole grid cooperates) ----
  // Fragment group (t, mts, kk, lane_s): 16 bf16 from two 32B runs of X.
  {
    const size_t ngroups = (size_t)kT * 2 * kKCX * 32;  // 1,048,576
    for (size_t gidx = (size_t)wg * kNT + tid; gidx < ngroups;
         gidx += (size_t)kNWG * kNT) {
      const int lane_s = (int)(gidx & 31);
      size_t g2 = gidx >> 5;
      const int kk  = (int)(g2 & (kKCX - 1)); g2 >>= 3;
      const int mts = (int)(g2 & 1);          g2 >>= 1;
      const int t   = (int)g2;
      const int b   = mts * 16 + (lane_s & 15);
      const int hi  = lane_s >> 4;
      const float* src = &X[((size_t)b * kT + t) * kD + kk * 32 + hi * 8];
      __bf16* dst = &xf[(((size_t)t * 2 + mts) * kKCX + kk) * kFragElems
                        + (size_t)lane_s * 16];
      #pragma unroll
      for (int e = 0; e < 8; ++e) dst[e]     = (__bf16)src[e];
      #pragma unroll
      for (int e = 0; e < 8; ++e) dst[8 + e] = (__bf16)src[16 + e];
    }
  }

  // ---- Phase 0b: weights -> LDS fragments, zero c state ----
  pack_weights(bW, Wx, Wh, wg, tid);
  for (int idx = tid; idx < 2 * 32 * 8; idx += kNT)
    (reinterpret_cast<float*>(cb))[idx] = 0.0f;

  const float biasv = bias[gq * kH + j0 + nloc];

  grid_barrier(bar, (unsigned)kNWG, tid);   // x fragments visible everywhere

  // h_new -> fragment-position constants for the combine stage (per thread).
  const int c_kkh = wg >> 1;                           // h-chunk this WG feeds
  const int c_e   = (nloc & 7) + ((wg & 1) << 3);      // element within lane
  const int c_lhi = ((nloc >> 3) & 1) << 4;            // lane bit4 from K-offset

  // ------------------------------ time scan --------------------------------
  for (int t = 0; t < kT; ++t) {
    const int cur = t & 1, nxt = cur ^ 1;
    const __bf16* __restrict__ xft = xf + ((size_t)t * 2 + mt) * kKCX * kFragElems;
    const __bf16* __restrict__ hfc = hf + ((size_t)cur * 2 + mt) * kKCH * kFragElems;

    // Hint next step's x fragments toward the WGP while we compute.
    if (gq == 0 && t + 1 < kT) {
      const __bf16* nx = xf + ((size_t)(t + 1) * 2 + mt) * kKCX * kFragElems
                         + (size_t)lane * 16;
      #pragma unroll
      for (int kk = 0; kk < kKCX; ++kk)
        __builtin_prefetch(nx + (size_t)kk * kFragElems, 0, 0);
    }

    // GEMM: each wave computes one 16x16 tile of z (gate gq, rows of mt).
    v8f_t acc;
    #pragma unroll
    for (int r = 0; r < 8; ++r) acc[r] = biasv;

    #pragma unroll
    for (int kk = 0; kk < kKC; ++kk) {
      const __bf16* asrc = (kk < kKCX)
          ? xft + (size_t)kk * kFragElems
          : hfc + (size_t)(kk - kKCX) * kFragElems;
      const v16bf_t af = *reinterpret_cast<const v16bf_t*>(asrc + (size_t)lane * 16);
      const v16bf_t bf = *reinterpret_cast<const v16bf_t*>(&bW[gq][kk][lane][0]);
      acc = __builtin_amdgcn_wmma_f32_16x16x32_bf16(
          false, af, false, bf, (short)0, acc, false, false);
    }

    *reinterpret_cast<v8f_t*>(&zb[wave][lane][0]) = acc;
    __syncthreads();

    // Gate combine + state update; write h_new straight into fragment layout.
    if (gq == 0) {
      __bf16* hdst = hf + (((size_t)nxt * 2 + mt) * kKCH + c_kkh) * kFragElems;
      #pragma unroll
      for (int r = 0; r < 8; ++r) {
        const float zi = zb[mt * 4 + 0][lane][r];
        const float zf = zb[mt * 4 + 1][lane][r];
        const float zg = zb[mt * 4 + 2][lane][r];
        const float zo = zb[mt * 4 + 3][lane][r];
        const float c_old = cb[mt][lane][r];
        const float c_new = sigf(zf) * c_old + sigf(zi) * tanh_fast(zg);
        const float h_new = sigf(zo) * tanh_fast(c_new);
        cb[mt][lane][r] = c_new;
        const int lane_dst = r + (hi16 << 3) + c_lhi;
        hdst[(size_t)lane_dst * 16 + c_e] = (__bf16)h_new;
      }
    }

    grid_barrier(bar, (unsigned)kNWG * (unsigned)(t + 2), tid);
  }

  // ---- Final dense: out = h_last @ Wd + bd (h_last in hf buf 0, T even) ---
  if (wave == 0) {
    const int mt2 = wg >> 4;
    const int o0  = (wg & 15) * 16;
    const __bf16* __restrict__ hl = hf + (size_t)mt2 * kKCH * kFragElems;

    v8f_t acc;
    const float bdv = bd[o0 + nloc];
    #pragma unroll
    for (int r = 0; r < 8; ++r) acc[r] = bdv;

    #pragma unroll
    for (int kk = 0; kk < kKCH; ++kk) {
      const v16bf_t af = *reinterpret_cast<const v16bf_t*>(
          hl + (size_t)kk * kFragElems + (size_t)lane * 16);
      const int n  = o0 + nloc;
      const int kb = kk * 32 + (hi16 ? 8 : 0);
      v16bf_t wf;
      #pragma unroll
      for (int e = 0; e < 16; ++e) wf[e] = (__bf16)Wd[(size_t)(kb + e + (e & 8)) * kO + n];
      acc = __builtin_amdgcn_wmma_f32_16x16x32_bf16(
          false, af, false, wf, (short)0, acc, false, false);
    }
    #pragma unroll
    for (int r = 0; r < 8; ++r) {
      const int b = mt2 * 16 + r + (hi16 << 3);
      out[(size_t)b * kO + o0 + nloc] = acc[r];
    }
  }
}

// ===========================================================================
// Fallback path: stage [x_t | h] through LDS each step (small ws).
// ===========================================================================
__global__ __launch_bounds__(kNT, 1)
void lstm_staged(const float* __restrict__ X, const float* __restrict__ Wx,
                 const float* __restrict__ Wh, const float* __restrict__ bias,
                 const float* __restrict__ Wd, const float* __restrict__ bd,
                 float* __restrict__ out, float* __restrict__ h0,
                 float* __restrict__ h1, unsigned* __restrict__ bar)
{
  extern __shared__ char smem[];
  __bf16 (*bW)[kKC][32][16] = reinterpret_cast<__bf16 (*)[kKC][32][16]>(smem + kS_OffBW);
  __bf16*  As               = reinterpret_cast<__bf16*>(smem + kS_OffAS);
  float  (*zb)[32][8]       = reinterpret_cast<float (*)[32][8]>(smem + kS_OffZB);
  float  (*cb)[32][8]       = reinterpret_cast<float (*)[32][8]>(smem + kS_OffCB);

  const int wg   = blockIdx.x;
  const int tid  = threadIdx.x;
  const int lane = tid & 31;
  const int wave = tid >> 5;
  const int mt   = wave >> 2;
  const int gq   = wave & 3;
  const int j0   = wg * 16;
  const int nloc = lane & 15;
  const int hi16 = (lane & 16) ? 1 : 0;

  pack_weights(bW, Wx, Wh, wg, tid);
  for (int idx = tid; idx < 2 * 32 * 8; idx += kNT)
    (reinterpret_cast<float*>(cb))[idx] = 0.0f;
  __syncthreads();

  const float biasv = bias[gq * kH + j0 + nloc];

  for (int t = 0; t < kT; ++t) {
    const float* __restrict__ hprev = (t & 1) ? h1 : h0;
    float*       __restrict__ hnext = (t & 1) ? h0 : h1;

    for (int idx = tid; idx < kB * kKtot; idx += kNT) {
      const int b = idx / kKtot;
      const int k = idx - b * kKtot;
      const float v = (k < kD) ? X[(size_t)b * kT * kD + (size_t)t * kD + k]
                               : hprev[b * kH + (k - kD)];
      As[b * kKPad + k] = (__bf16)v;
    }
    __syncthreads();

    v8f_t acc;
    #pragma unroll
    for (int r = 0; r < 8; ++r) acc[r] = biasv;

    const int m = mt * 16 + nloc;
    #pragma unroll 4
    for (int kk = 0; kk < kKC; ++kk) {
      const int ka = kk * 32 + (hi16 ? 8 : 0);
      v16bf_t af;
      #pragma unroll
      for (int e = 0; e < 8; ++e) af[e]     = As[m * kKPad + ka + e];
      #pragma unroll
      for (int e = 0; e < 8; ++e) af[8 + e] = As[m * kKPad + ka + 16 + e];
      const v16bf_t bf = *reinterpret_cast<const v16bf_t*>(&bW[gq][kk][lane][0]);
      acc = __builtin_amdgcn_wmma_f32_16x16x32_bf16(
          false, af, false, bf, (short)0, acc, false, false);
    }

    *reinterpret_cast<v8f_t*>(&zb[wave][lane][0]) = acc;
    __syncthreads();

    if (gq == 0) {
      const int col = j0 + nloc;
      #pragma unroll
      for (int r = 0; r < 8; ++r) {
        const float zi = zb[mt * 4 + 0][lane][r];
        const float zf = zb[mt * 4 + 1][lane][r];
        const float zg = zb[mt * 4 + 2][lane][r];
        const float zo = zb[mt * 4 + 3][lane][r];
        const float c_old = cb[mt][lane][r];
        const float c_new = sigf(zf) * c_old + sigf(zi) * tanh_fast(zg);
        const float h_new = sigf(zo) * tanh_fast(c_new);
        cb[mt][lane][r] = c_new;
        const int b = mt * 16 + r + (hi16 << 3);
        hnext[b * kH + col] = h_new;
      }
    }

    grid_barrier(bar, (unsigned)kNWG * (unsigned)(t + 1), tid);
  }

  if (wave == 0) {
    const int mt2 = wg >> 4;
    const int o0  = (wg & 15) * 16;
    const float* __restrict__ hl = h0;

    v8f_t acc;
    const float bdv = bd[o0 + nloc];
    #pragma unroll
    for (int r = 0; r < 8; ++r) acc[r] = bdv;

    const int m = mt2 * 16 + nloc;
    #pragma unroll 4
    for (int kk = 0; kk < kKCH; ++kk) {
      const int ka = kk * 32 + (hi16 ? 8 : 0);
      v16bf_t af;
      #pragma unroll
      for (int e = 0; e < 8; ++e) af[e]     = (__bf16)hl[m * kH + ka + e];
      #pragma unroll
      for (int e = 0; e < 8; ++e) af[8 + e] = (__bf16)hl[m * kH + ka + 16 + e];
      const int n  = o0 + nloc;
      const int kb = kk * 32 + (hi16 ? 8 : 0);
      v16bf_t wf;
      #pragma unroll
      for (int e = 0; e < 16; ++e) wf[e] = (__bf16)Wd[(size_t)(kb + e + (e & 8)) * kO + n];
      acc = __builtin_amdgcn_wmma_f32_16x16x32_bf16(
          false, af, false, wf, (short)0, acc, false, false);
    }
    #pragma unroll
    for (int r = 0; r < 8; ++r) {
      const int b = mt2 * 16 + r + (hi16 << 3);
      out[(size_t)b * kO + o0 + nloc] = acc[r];
    }
  }
}

// ---------------------------------------------------------------------------
extern "C" void kernel_launch(void* const* d_in, const int* in_sizes, int n_in,
                              void* d_out, int out_size, void* d_ws, size_t ws_size,
                              hipStream_t stream) {
  (void)in_sizes; (void)n_in; (void)out_size;

  const float* X    = (const float*)d_in[0];
  const float* Wx   = (const float*)d_in[1];
  const float* Wh   = (const float*)d_in[2];
  const float* bias = (const float*)d_in[3];
  const float* Wd   = (const float*)d_in[4];
  const float* bd   = (const float*)d_in[5];
  float*       out  = (float*)d_out;
  char*        ws   = (char*)d_ws;

  if (ws_size >= kNeedFrag) {
    unsigned* bar = (unsigned*)ws;
    __bf16*   hf  = (__bf16*)(ws + kOffHF);
    __bf16*   xf  = (__bf16*)(ws + kOffXF);
    // Zero barrier counter + h fragment buffers (xf fully overwritten).
    hipMemsetAsync(ws, 0, kOffXF, stream);
    hipFuncSetAttribute(reinterpret_cast<const void*>(lstm_frag),
                        hipFuncAttributeMaxDynamicSharedMemorySize, (int)kF_Smem);
    lstm_frag<<<dim3(kNWG), dim3(kNT), kF_Smem, stream>>>(
        X, Wx, Wh, bias, Wd, bd, out, xf, hf, bar);
  } else if (ws_size >= kNeedStaged) {
    const size_t hbytes = (size_t)kB * kH * sizeof(float);
    unsigned* bar = (unsigned*)ws;
    float*    h0  = (float*)(ws + 256);
    float*    h1  = (float*)(ws + 256 + hbytes);
    hipMemsetAsync(ws, 0, 256 + 2 * hbytes, stream);
    hipFuncSetAttribute(reinterpret_cast<const void*>(lstm_staged),
                        hipFuncAttributeMaxDynamicSharedMemorySize, (int)kS_Smem);
    lstm_staged<<<dim3(kNWG), dim3(kNT), kS_Smem, stream>>>(
        X, Wx, Wh, bias, Wd, bd, out, h0, h1, bar);
  }
}
